// MiniLLMBlock_9380208574859
// MI455X (gfx1250) — matrix operand order repaired and checked by reading, not compiled, not run on hardware
//
#include <hip/hip_runtime.h>
#include <hip/hip_bf16.h>

// ---------------------------------------------------------------------------
// MiniLLMBlock for MI455X (gfx1250): all GEMM-like work on v_wmma_f32_16x16x32_bf16
// wave32, WGP, 8-wave (256-thread) workgroups. Compute-bound (~0.48 TFLOP vs
// ~8us of HBM traffic at 23.3TB/s), so everything is pushed through bf16 WMMA.
// v2: double-buffered single-barrier GEMM; attention keeps K, V and P all in
// LDS (Ks/Ps aliased) so the P@V WMMAs feed from ds_load instead of per-wmma
// serialized global loads.
// ---------------------------------------------------------------------------

#define HIDDEN   2048
#define HEADS    16
#define HEAD_DIM 128
#define BATCH    2
#define SEQ      2048
#define ROWS     (BATCH * SEQ)   // 4096

typedef __attribute__((ext_vector_type(16))) __bf16 v16bf;
typedef __attribute__((ext_vector_type(8)))  float  v8f;

union Frag16 { v16bf v; uint4 q[2]; };

__device__ __forceinline__ unsigned short f2bf(float f) {
    union { float f; unsigned int u; } c; c.f = f;
    unsigned int u = c.u;
    if ((u & 0x7fffffffu) > 0x7f800000u)          // NaN
        return (unsigned short)((u >> 16) | 0x0040u);
    return (unsigned short)((u + 0x7fffu + ((u >> 16) & 1u)) >> 16);  // RNE
}

__device__ __forceinline__ v8f wmma_bf16(const Frag16& a, const Frag16& b, v8f c) {
    // D = A(16x32 bf16) * B(32x16 bf16) + C(16x16 f32)
    return __builtin_amdgcn_wmma_f32_16x16x32_bf16(
        false, a.v, false, b.v, (short)0, c, false, false);
}

// ---------------------------------------------------------------------------
// fp32 -> bf16 weight convert + transpose:  Wt[n*K + k] = bf16(W[k*N + n])
// ---------------------------------------------------------------------------
__global__ void convtr_kernel(const float* __restrict__ W,
                              unsigned short* __restrict__ Wt, int K, int N) {
    const size_t idx = (size_t)blockIdx.x * blockDim.x + threadIdx.x;
    if (idx >= (size_t)K * N) return;
    const int k = (int)(idx % K);
    const int n = (int)(idx / K);
    Wt[idx] = f2bf(W[(size_t)k * N + n]);
}

// ---------------------------------------------------------------------------
// LayerNorm: one 256-thread block per row of 2048, bf16 output
// ---------------------------------------------------------------------------
__global__ __launch_bounds__(256)
void layernorm_kernel(const float* __restrict__ x, const float* __restrict__ gamma,
                      const float* __restrict__ beta, unsigned short* __restrict__ out) {
    const int row = blockIdx.x;
    const int tid = threadIdx.x;
    const float* xr = x + (size_t)row * HIDDEN;
    float vals[8];
    float s = 0.f, s2 = 0.f;
#pragma unroll
    for (int i = 0; i < 8; ++i) {
        float v = xr[tid + (i << 8)];
        vals[i] = v; s += v; s2 += v * v;
    }
#pragma unroll
    for (int off = 16; off > 0; off >>= 1) {
        s  += __shfl_xor(s,  off, 32);
        s2 += __shfl_xor(s2, off, 32);
    }
    __shared__ float rs[8], rs2[8];
    if ((tid & 31) == 0) { rs[tid >> 5] = s; rs2[tid >> 5] = s2; }
    __syncthreads();
    s = 0.f; s2 = 0.f;
#pragma unroll
    for (int w = 0; w < 8; ++w) { s += rs[w]; s2 += rs2[w]; }
    const float mean = s * (1.0f / HIDDEN);
    const float var  = s2 * (1.0f / HIDDEN) - mean * mean;
    const float rstd = rsqrtf(var + 1e-5f);
    unsigned short* orow = out + (size_t)row * HIDDEN;
#pragma unroll
    for (int i = 0; i < 8; ++i) {
        const int c = tid + (i << 8);
        orow[c] = f2bf((vals[i] - mean) * rstd * gamma[c] + beta[c]);
    }
}

// ---------------------------------------------------------------------------
// GEMM: C[M,N] = epi( A_bf16[M,K] @ Bt_bf16[N,K] )   (Bt is W transposed)
// 128x128x32 tile / block, 8 waves, each wave = 64x32 output (4x2 wmma tiles).
// Ping-pong LDS buffers -> one s_barrier per k-step; tile kb is stored into
// buf[kb&1], whose last reader (iter kb-2) is fenced by iter kb-1's barrier.
// ---------------------------------------------------------------------------
#define TILE_M 128
#define TILE_N 128
#define TILE_K 32

template<bool HAS_BIAS, bool SILU, bool HAS_RES, bool OUT_BF16>
__global__ __launch_bounds__(256)
void gemm_kernel(const unsigned short* __restrict__ A,
                 const unsigned short* __restrict__ Bt,
                 const float* __restrict__ bias,
                 const float* __restrict__ residual,
                 void* __restrict__ outv,
                 int M, int N, int K) {
    __shared__ unsigned short As[2][TILE_M][TILE_K];   // 16 KB
    __shared__ unsigned short Bs[2][TILE_N][TILE_K];   // 16 KB
    const int tid  = threadIdx.x;
    const int lane = tid & 31;
    const int wave = tid >> 5;
    const int l16  = lane & 15;
    const int hi   = (lane >> 4) & 1;
    const int wr   = wave >> 2;              // 0..1 -> 64 rows
    const int wc   = wave & 3;               // 0..3 -> 32 cols
    const int bm0  = blockIdx.y * TILE_M;
    const int bn0  = blockIdx.x * TILE_N;
    const int srow = tid >> 1;               // staging: 2 threads/row, 16 bf16 each
    const int scol = (tid & 1) << 4;

    v8f vzero = {0.f,0.f,0.f,0.f,0.f,0.f,0.f,0.f};
    v8f acc[4][2];
#pragma unroll
    for (int i = 0; i < 4; ++i)
#pragma unroll
        for (int j = 0; j < 2; ++j) acc[i][j] = vzero;

    const unsigned short* aptr = A  + (size_t)(bm0 + srow) * K + scol;
    const unsigned short* bptr = Bt + (size_t)(bn0 + srow) * K + scol;
    uint4 ra0 = *(const uint4*)aptr;
    uint4 ra1 = *(const uint4*)(aptr + 8);
    uint4 rb0 = *(const uint4*)bptr;
    uint4 rb1 = *(const uint4*)(bptr + 8);

    const int nk = K / TILE_K;
    for (int kb = 0; kb < nk; ++kb) {
        const int pb = kb & 1;
        // store tile kb (regs were loaded last iteration / preloaded)
        *(uint4*)&As[pb][srow][scol]     = ra0;
        *(uint4*)&As[pb][srow][scol + 8] = ra1;
        *(uint4*)&Bs[pb][srow][scol]     = rb0;
        *(uint4*)&Bs[pb][srow][scol + 8] = rb1;
        if (kb + 1 < nk) {                    // fetch tile kb+1, overlaps compute
            aptr += TILE_K; bptr += TILE_K;
            ra0 = *(const uint4*)aptr; ra1 = *(const uint4*)(aptr + 8);
            rb0 = *(const uint4*)bptr; rb1 = *(const uint4*)(bptr + 8);
            if (kb + 2 < nk) {                // gfx1250 global_prefetch_b8
                __builtin_prefetch(aptr + TILE_K, 0, 0);
                __builtin_prefetch(bptr + TILE_K, 0, 0);
            }
        }
        __syncthreads();
        Frag16 af[4], bfm[2];
#pragma unroll
        for (int i = 0; i < 4; ++i) {         // A layout: two 8-elem K runs
            const unsigned short* ap = &As[pb][wr * 64 + i * 16 + l16][0];
            af[i].q[0] = *(const uint4*)(ap + (hi ? 8 : 0));
            af[i].q[1] = *(const uint4*)(ap + (hi ? 24 : 16));
        }
#pragma unroll
        for (int j = 0; j < 2; ++j) {         // B layout: two 8-elem K runs
            const unsigned short* bp = &Bs[pb][wc * 32 + j * 16 + l16][0];
            bfm[j].q[0] = *(const uint4*)(bp + (hi ? 8 : 0));
            bfm[j].q[1] = *(const uint4*)(bp + (hi ? 24 : 16));
        }
#pragma unroll
        for (int i = 0; i < 4; ++i)
#pragma unroll
            for (int j = 0; j < 2; ++j)
                acc[i][j] = wmma_bf16(af[i], bfm[j], acc[i][j]);
    }

    // epilogue: C layout -> row = base + (hi? g+8 : g), col = base + l16
#pragma unroll
    for (int i = 0; i < 4; ++i) {
#pragma unroll
        for (int j = 0; j < 2; ++j) {
            const int n = bn0 + wc * 32 + j * 16 + l16;
            const float bv = HAS_BIAS ? bias[n] : 0.0f;
#pragma unroll
            for (int g = 0; g < 8; ++g) {
                const int m = bm0 + wr * 64 + i * 16 + (hi ? g + 8 : g);
                float v = acc[i][j][g] + bv;
                if (SILU)    v = v / (1.0f + __expf(-v));
                if (HAS_RES) v += residual[(size_t)m * N + n];
                if (OUT_BF16) ((unsigned short*)outv)[(size_t)m * N + n] = f2bf(v);
                else          ((float*)outv)[(size_t)m * N + n] = v;
            }
        }
    }
}

// ---------------------------------------------------------------------------
// Vt[bh][d][t] = V[b*SEQ + t][h*HEAD_DIM + d]   (bf16 -> bf16)
// ---------------------------------------------------------------------------
__global__ void transpose_v_kernel(const unsigned short* __restrict__ V,
                                   unsigned short* __restrict__ Vt) {
    const size_t idx = (size_t)blockIdx.x * blockDim.x + threadIdx.x;
    const int t   = (int)(idx & (SEQ - 1));
    const size_t r1 = idx >> 11;
    const int d   = (int)(r1 & (HEAD_DIM - 1));
    const size_t bh = r1 >> 7;            // 0..31
    const int b = (int)(bh >> 4), h = (int)(bh & 15);
    Vt[idx] = V[((size_t)(b * SEQ + t)) * HIDDEN + h * HEAD_DIM + d];
}

// ---------------------------------------------------------------------------
// Causal flash attention. Block = (q-tile of 128, one (b,h)); 8 waves, each
// owning 16 query rows so online softmax is wave-local.
// LDS (64 KB total): KP[128][128] holds the K tile during S=QK^T, then is
// reused (after a barrier) as the P tile for the C-layout -> A-layout
// round-trip; Vs[128][128] holds the V tile [d][t] so all P@V operands come
// from LDS (each K/V tile hits HBM once per block instead of 8x).
// ---------------------------------------------------------------------------
__global__ __launch_bounds__(256)
void attn_kernel(const unsigned short* __restrict__ Q,
                 const unsigned short* __restrict__ Kc,
                 const unsigned short* __restrict__ Vt,
                 unsigned short* __restrict__ AO) {
    __shared__ unsigned short KP[128][HEAD_DIM];   // 32 KB: K tile, then P tile
    __shared__ unsigned short Vs[128][128];        // 32 KB: V tile [d][t]
    const int tid  = threadIdx.x;
    const int lane = tid & 31;
    const int wave = tid >> 5;
    const int l16  = lane & 15;
    const int hi   = (lane >> 4) & 1;
    const int bh   = blockIdx.y;
    const int b    = bh >> 4, h = bh & 15;
    const int q0   = blockIdx.x * 128;
    const int m0   = q0 + wave * 16;               // this wave's 16 query rows

    // Q A-fragments for k(d) = 0..127 : 4 frags, loaded once
    Frag16 qf[4];
    const unsigned short* qp =
        Q + ((size_t)(b * SEQ + m0 + l16)) * HIDDEN + h * HEAD_DIM;
#pragma unroll
    for (int s = 0; s < 4; ++s) {
        qf[s].q[0] = *(const uint4*)(qp + s * 32 + (hi ? 8 : 0));
        qf[s].q[1] = *(const uint4*)(qp + s * 32 + (hi ? 24 : 16));
    }

    v8f vzero = {0.f,0.f,0.f,0.f,0.f,0.f,0.f,0.f};
    float mrun[8], lrun[8];
    v8f oacc[8];
#pragma unroll
    for (int g = 0; g < 8; ++g) { mrun[g] = -1e30f; lrun[g] = 0.0f; }
#pragma unroll
    for (int j = 0; j < 8; ++j) oacc[j] = vzero;

    const float scale = 0.0883883476483184405f;    // 1/sqrt(128)
    const unsigned short* vbase = Vt + (size_t)bh * HEAD_DIM * SEQ;
    const int tend = q0 + 128;                     // causal: keys < q-tile end

    for (int t0 = 0; t0 < tend; t0 += 128) {
        __syncthreads();   // previous iteration's P@V reads of KP/Vs complete
        {   // stage K tile (rows t0..t0+127) and V tile (d-major), 128B each/thread
            const int r = tid >> 1;
            const int c = (tid & 1) * 64;
            const uint4* ksrc = (const uint4*)(Kc +
                ((size_t)(b * SEQ + t0 + r)) * HIDDEN + h * HEAD_DIM + c);
            const uint4* vsrc = (const uint4*)(vbase + (size_t)r * SEQ + t0 + c);
            uint4* kdst = (uint4*)&KP[r][c];
            uint4* vdst = (uint4*)&Vs[r][c];
#pragma unroll
            for (int q = 0; q < 8; ++q) kdst[q] = ksrc[q];
#pragma unroll
            for (int q = 0; q < 8; ++q) vdst[q] = vsrc[q];
        }
        __syncthreads();

        // S = Q K^T  (16 rows x 128 keys per wave); batch 8 B-frags per k-step
        v8f sacc[8];
#pragma unroll
        for (int j = 0; j < 8; ++j) sacc[j] = vzero;
#pragma unroll
        for (int ss = 0; ss < 4; ++ss) {
            Frag16 kf[8];
#pragma unroll
            for (int j = 0; j < 8; ++j) {
                const unsigned short* kp = &KP[j * 16 + l16][ss * 32];
                kf[j].q[0] = *(const uint4*)(kp + (hi ? 8 : 0));
                kf[j].q[1] = *(const uint4*)(kp + (hi ? 24 : 16));
            }
#pragma unroll
            for (int j = 0; j < 8; ++j)
                sacc[j] = wmma_bf16(qf[ss], kf[j], sacc[j]);
        }

        // scale + causal mask + row max
        float rmax[8];
#pragma unroll
        for (int g = 0; g < 8; ++g) rmax[g] = -1e30f;
#pragma unroll
        for (int j = 0; j < 8; ++j) {
            const int col = t0 + j * 16 + l16;
#pragma unroll
            for (int g = 0; g < 8; ++g) {
                const int row = m0 + (hi ? g + 8 : g);
                float v = sacc[j][g] * scale;
                if (col > row) v = -1e30f;
                sacc[j][g] = v;
                rmax[g] = fmaxf(rmax[g], v);
            }
        }
#pragma unroll
        for (int g = 0; g < 8; ++g) {       // reduce across the 16-lane half
            rmax[g] = fmaxf(rmax[g], __shfl_xor(rmax[g], 1, 16));
            rmax[g] = fmaxf(rmax[g], __shfl_xor(rmax[g], 2, 16));
            rmax[g] = fmaxf(rmax[g], __shfl_xor(rmax[g], 4, 16));
            rmax[g] = fmaxf(rmax[g], __shfl_xor(rmax[g], 8, 16));
        }
        float corr[8], rsum[8];
#pragma unroll
        for (int g = 0; g < 8; ++g) {
            const float mn = fmaxf(mrun[g], rmax[g]);
            corr[g] = __expf(mrun[g] - mn);
            mrun[g] = mn;
            rsum[g] = 0.0f;
        }
#pragma unroll
        for (int j = 0; j < 8; ++j)
#pragma unroll
            for (int g = 0; g < 8; ++g) {
                const float p = __expf(sacc[j][g] - mrun[g]);
                sacc[j][g] = p;
                rsum[g] += p;
            }
#pragma unroll
        for (int g = 0; g < 8; ++g) {
            rsum[g] += __shfl_xor(rsum[g], 1, 16);
            rsum[g] += __shfl_xor(rsum[g], 2, 16);
            rsum[g] += __shfl_xor(rsum[g], 4, 16);
            rsum[g] += __shfl_xor(rsum[g], 8, 16);
            lrun[g] = lrun[g] * corr[g] + rsum[g];
        }
#pragma unroll
        for (int j = 0; j < 8; ++j)
#pragma unroll
            for (int g = 0; g < 8; ++g) oacc[j][g] *= corr[g];

        __syncthreads();   // all waves done reading K tile; KP becomes P tile

        // P (C-layout) -> own 16 rows of KP (per-wave region, in-order DS)
#pragma unroll
        for (int j = 0; j < 8; ++j)
#pragma unroll
            for (int g = 0; g < 8; ++g)
                KP[wave * 16 + (hi ? g + 8 : g)][j * 16 + l16] = f2bf(sacc[j][g]);

        // O += P @ V  (all operands from LDS; batch 8 V-frags per k-step)
#pragma unroll
        for (int ks = 0; ks < 4; ++ks) {
            Frag16 pf;
            const unsigned short* pp = &KP[wave * 16 + l16][ks * 32];
            pf.q[0] = *(const uint4*)(pp + (hi ? 8 : 0));
            pf.q[1] = *(const uint4*)(pp + (hi ? 24 : 16));
            Frag16 vf[8];
#pragma unroll
            for (int jd = 0; jd < 8; ++jd) {
                const unsigned short* vp = &Vs[jd * 16 + l16][ks * 32];
                vf[jd].q[0] = *(const uint4*)(vp + (hi ? 8 : 0));
                vf[jd].q[1] = *(const uint4*)(vp + (hi ? 24 : 16));
            }
#pragma unroll
            for (int jd = 0; jd < 8; ++jd)
                oacc[jd] = wmma_bf16(pf, vf[jd], oacc[jd]);
        }
    }

    // normalize and scatter to merged-head layout [B*T][HIDDEN] (bf16)
#pragma unroll
    for (int jd = 0; jd < 8; ++jd) {
#pragma unroll
        for (int g = 0; g < 8; ++g) {
            const int row = m0 + (hi ? g + 8 : g);
            const int d   = jd * 16 + l16;
            const float v = oacc[jd][g] / lrun[g];
            AO[((size_t)(b * SEQ) + row) * HIDDEN + h * HEAD_DIM + d] = f2bf(v);
        }
    }
}

// ---------------------------------------------------------------------------
extern "C" void kernel_launch(void* const* d_in, const int* in_sizes, int n_in,
                              void* d_out, int out_size, void* d_ws, size_t ws_size,
                              hipStream_t stream) {
    const float* x   = (const float*)d_in[0];
    // d_in[1]: causal mask — implicit in attn_kernel
    const float* Wq  = (const float*)d_in[2];
    const float* Wk  = (const float*)d_in[3];
    const float* Wv  = (const float*)d_in[4];
    const float* Wo  = (const float*)d_in[5];
    const float* g1  = (const float*)d_in[6];
    const float* bn1 = (const float*)d_in[7];
    const float* g2w = (const float*)d_in[8];
    const float* bn2 = (const float*)d_in[9];
    const float* W1  = (const float*)d_in[10];
    const float* b1  = (const float*)d_in[11];
    const float* W2  = (const float*)d_in[12];
    const float* b2  = (const float*)d_in[13];
    float* out = (float*)d_out;

    char* ws = (char*)d_ws;
    size_t off = 0;
    auto alloc = [&](size_t bytes) -> unsigned short* {
        unsigned short* p = (unsigned short*)(ws + off);
        off += (bytes + 255) & ~(size_t)255;
        return p;
    };
    const size_t SQ_B  = (size_t)HIDDEN * HIDDEN * 2;        //  8 MB
    const size_t MLP_B = (size_t)HIDDEN * 4 * HIDDEN * 2;    // 32 MB
    const size_t ACT_B = (size_t)ROWS * HIDDEN * 2;          // 16 MB

    unsigned short* Wq_t = alloc(SQ_B);
    unsigned short* Wk_t = alloc(SQ_B);
    unsigned short* Wv_t = alloc(SQ_B);
    unsigned short* Wo_t = alloc(SQ_B);
    unsigned short* W1_t = alloc(MLP_B);
    unsigned short* W2_t = alloc(MLP_B);
    unsigned short* hb   = alloc(ACT_B);
    unsigned short* Qb   = alloc(ACT_B);
    unsigned short* Kb   = alloc(ACT_B);
    unsigned short* Vb   = alloc(ACT_B);
    unsigned short* Vtb  = alloc(ACT_B);
    unsigned short* AOb  = alloc(ACT_B);
    unsigned short* h2b  = alloc(ACT_B);
    unsigned short* act  = alloc((size_t)ROWS * 4 * HIDDEN * 2);  // 64 MB

    const int TPB = 256;
    // weight convert + transpose
    convtr_kernel<<<(HIDDEN * HIDDEN) / TPB, TPB, 0, stream>>>(Wq, Wq_t, HIDDEN, HIDDEN);
    convtr_kernel<<<(HIDDEN * HIDDEN) / TPB, TPB, 0, stream>>>(Wk, Wk_t, HIDDEN, HIDDEN);
    convtr_kernel<<<(HIDDEN * HIDDEN) / TPB, TPB, 0, stream>>>(Wv, Wv_t, HIDDEN, HIDDEN);
    convtr_kernel<<<(HIDDEN * HIDDEN) / TPB, TPB, 0, stream>>>(Wo, Wo_t, HIDDEN, HIDDEN);
    convtr_kernel<<<(HIDDEN * 4 * HIDDEN) / TPB, TPB, 0, stream>>>(W1, W1_t, HIDDEN, 4 * HIDDEN);
    convtr_kernel<<<(HIDDEN * 4 * HIDDEN) / TPB, TPB, 0, stream>>>(W2, W2_t, 4 * HIDDEN, HIDDEN);

    // LN1 -> h (bf16)
    layernorm_kernel<<<ROWS, TPB, 0, stream>>>(x, g1, bn1, hb);

    // Q,K,V projections
    dim3 gN2048(HIDDEN / TILE_N, ROWS / TILE_M);          // (16,32)
    gemm_kernel<false,false,false,true><<<gN2048, TPB, 0, stream>>>(hb, Wq_t, nullptr, nullptr, Qb, ROWS, HIDDEN, HIDDEN);
    gemm_kernel<false,false,false,true><<<gN2048, TPB, 0, stream>>>(hb, Wk_t, nullptr, nullptr, Kb, ROWS, HIDDEN, HIDDEN);
    gemm_kernel<false,false,false,true><<<gN2048, TPB, 0, stream>>>(hb, Wv_t, nullptr, nullptr, Vb, ROWS, HIDDEN, HIDDEN);

    // V -> Vt[bh][d][t]
    transpose_v_kernel<<<(ROWS * HIDDEN) / TPB, TPB, 0, stream>>>(Vb, Vtb);

    // causal flash attention
    attn_kernel<<<dim3(SEQ / 128, BATCH * HEADS), TPB, 0, stream>>>(Qb, Kb, Vtb, AOb);

    // x + AO @ Wo -> d_out (fp32)
    gemm_kernel<false,false,true,false><<<gN2048, TPB, 0, stream>>>(AOb, Wo_t, nullptr, x, out, ROWS, HIDDEN, HIDDEN);

    // LN2 -> h2 (bf16)
    layernorm_kernel<<<ROWS, TPB, 0, stream>>>(out, g2w, bn2, h2b);

    // MLP: silu(h2 @ W1 + b1) -> act (bf16), then d_out += act @ W2 + b2
    dim3 gN8192((4 * HIDDEN) / TILE_N, ROWS / TILE_M);    // (64,32)
    gemm_kernel<true,true,false,true><<<gN8192, TPB, 0, stream>>>(h2b, W1_t, b1, nullptr, act, ROWS, 4 * HIDDEN, HIDDEN);
    gemm_kernel<true,false,true,false><<<gN2048, TPB, 0, stream>>>(act, W2_t, b2, out, out, ROWS, HIDDEN, 4 * HIDDEN);
}
